// Zamba2MambaDecoderLayer_50448685859255
// MI455X (gfx1250) — matrix operand order repaired and checked by reading, not compiled, not run on hardware
//
#include <hip/hip_runtime.h>
#include <stdint.h>

// ---------------- Problem constants (from reference) ----------------
#define D_MODEL   2048
#define D_INNER   4096
#define D_STATE   64
#define HEADDIM   64
#define NHEADS    64
#define D_CONV    4
#define CONV_DIM  (D_INNER + 2 * D_STATE)                 // 4224
#define E_TOTAL   (2 * D_INNER + 2 * D_STATE + NHEADS)    // 8384
#define BATCH     2
#define SEQ       2048
#define NTOK      (BATCH * SEQ)                           // 4096
#define EPS_RMS   1e-5f
#define NCHUNK    16                                      // chunks per sequence
#define CHUNKL    (SEQ / NCHUNK)                          // 128 steps per chunk
#define STATE_SZ  (HEADDIM * D_STATE)                     // 4096 floats per (b,h) state

// ---------------- Vector / WMMA fragment types ----------------
typedef __attribute__((ext_vector_type(16))) __bf16        v16bf;
typedef __attribute__((ext_vector_type(8)))  float         v8f;
typedef __attribute__((ext_vector_type(4)))  unsigned int  v4u;
typedef __attribute__((ext_vector_type(4)))  float         v4f;

union Frag { v16bf v; v4u u[2]; };   // 32 bytes = 8 VGPRs (wave32)

// ---------------- Small device helpers ----------------
__device__ __forceinline__ unsigned short f32_to_bf16_bits(float f) {
    union { float f; uint32_t u; } in; in.f = f;
    uint32_t r = in.u + 0x7FFFu + ((in.u >> 16) & 1u);   // round-to-nearest-even
    return (unsigned short)(r >> 16);
}
__device__ __forceinline__ float sigmoid_(float x) { return 1.0f / (1.0f + __expf(-x)); }
__device__ __forceinline__ float silu_(float x)    { return x * sigmoid_(x); }
__device__ __forceinline__ float softplus_(float x){ return (x > 20.0f) ? x : log1pf(__expf(x)); }

// =====================================================================
// Kernel 1: RMSNorm(hidden) -> bf16 u    (one block per token row)
// =====================================================================
__global__ void __launch_bounds__(256)
rmsnorm_bf16_kernel(const float* __restrict__ x, const float* __restrict__ w,
                    unsigned short* __restrict__ out, int d, float eps) {
    __shared__ float red[256];
    const size_t row = blockIdx.x;
    const float* xr = x + row * (size_t)d;
    float ss = 0.0f;
    for (int i = threadIdx.x; i < d; i += 256) { float v = xr[i]; ss += v * v; }
    red[threadIdx.x] = ss;
    __syncthreads();
    for (int s = 128; s > 0; s >>= 1) {
        if ((int)threadIdx.x < s) red[threadIdx.x] += red[threadIdx.x + s];
        __syncthreads();
    }
    const float scale = rsqrtf(red[0] / (float)d + eps);
    unsigned short* orow = out + row * (size_t)d;
    for (int i = threadIdx.x; i < d; i += 256)
        orow[i] = f32_to_bf16_bits(xr[i] * scale * w[i]);
}

// =====================================================================
// Kernel 2: fp32 -> bf16 elementwise (weight down-convert)
// =====================================================================
__global__ void __launch_bounds__(256)
f32_to_bf16_kernel(const float* __restrict__ in, unsigned short* __restrict__ out, size_t n) {
    size_t i = (size_t)blockIdx.x * 256 + threadIdx.x;
    if (i < n) out[i] = f32_to_bf16_bits(in[i]);
}

// =====================================================================
// Kernel 3: bf16 WMMA GEMM   C[M][N] = A[M][K] * B[N][K]^T (+ residual)
// One wave computes a 64x32 tile via 4x2 v_wmma_f32_16x16x32_bf16 accs.
// Fragment addressing (A and B share a contiguous k order; C/D per CDNA5 ISA 7.12.2):
//   A 16x32 bf16 : row=lane&15, K = 16*(lane>=16) + [0..15]  (32 contiguous bytes)
//   B 32x16 bf16 : col=lane&15, K = 16*(lane>=16) + [0..15]  (32 contiguous bytes)
//   C/D 16x16 f32: vgpr r -> M = r + 8*(lane>=16), N = lane&15
// =====================================================================
__global__ void __launch_bounds__(256)
gemm_bf16_wmma(const unsigned short* __restrict__ A,
               const unsigned short* __restrict__ B,
               const float* __restrict__ residual,
               float* __restrict__ C,
               int M, int N, int K) {
    const int lane = threadIdx.x & 31;
    const int wid  = (int)blockIdx.x * 8 + ((int)threadIdx.x >> 5);
    const int ntile = (N + 31) >> 5;              // 32-col wave tiles
    const int mt = wid / ntile;
    const int nt = wid - mt * ntile;
    const int mbase = mt << 6;                    // 64 rows per wave tile
    const int nbase = nt << 5;
    if (mbase >= M) return;

    const int lh = lane >> 4;                     // 0 / 1 half-wave
    const int ll = lane & 15;

    v8f acc[4][2] = {};

    int arow[4];
#pragma unroll
    for (int i = 0; i < 4; ++i) {
        int r = mbase + i * 16 + ll;
        arow[i] = (r < M) ? r : (M - 1);
    }
    int bcol[2];
#pragma unroll
    for (int j = 0; j < 2; ++j) {
        int c = nbase + j * 16 + ll;
        bcol[j] = (c < N) ? c : (N - 1);
    }

    for (int k0 = 0; k0 < K; k0 += 32) {
        Frag a[4], b[2];
        const int ak = k0 + (lh << 4);
#pragma unroll
        for (int i = 0; i < 4; ++i) {
            const unsigned short* ap = A + (size_t)arow[i] * K + ak;
            a[i].u[0] = *(const v4u*)(ap);        // K contiguous 0..7
            a[i].u[1] = *(const v4u*)(ap + 8);    // K contiguous 8..15
        }
        const int bk = k0 + (lh << 4);
#pragma unroll
        for (int j = 0; j < 2; ++j) {
            const unsigned short* bp = B + (size_t)bcol[j] * K + bk;
            b[j].u[0] = *(const v4u*)(bp);        // K contiguous 0..7
            b[j].u[1] = *(const v4u*)(bp + 8);    // K contiguous 8..15
        }
        if (k0 + 32 < K) {                        // gfx1250 global_prefetch_b8
            __builtin_prefetch(A + (size_t)arow[0] * K + ak + 32, 0, 3);
            __builtin_prefetch(B + (size_t)bcol[0] * K + bk + 32, 0, 3);
        }
#pragma unroll
        for (int i = 0; i < 4; ++i)
#pragma unroll
            for (int j = 0; j < 2; ++j)
                acc[i][j] = __builtin_amdgcn_wmma_f32_16x16x32_bf16(
                    false, a[i].v, false, b[j].v, (short)0, acc[i][j], false, false);
    }

#pragma unroll
    for (int i = 0; i < 4; ++i) {
        const int m0 = mbase + i * 16 + (lh << 3);
#pragma unroll
        for (int j = 0; j < 2; ++j) {
            const int n = nbase + j * 16 + ll;
            if (n >= N) continue;
#pragma unroll
            for (int r = 0; r < 8; ++r) {
                const int m = m0 + r;
                if (m >= M) continue;
                const size_t idx = (size_t)m * N + n;
                float v = acc[i][j][r];
                if (residual) v += residual[idx];
                C[idx] = v;
            }
        }
    }
}

// =====================================================================
// Kernel 4: depthwise causal conv1d (k=4) + bias + SiLU over xBC slice
// =====================================================================
__global__ void __launch_bounds__(256)
conv_silu_kernel(const float* __restrict__ zx, const float* __restrict__ cw,
                 const float* __restrict__ cb, float* __restrict__ out, size_t total) {
    size_t idx = (size_t)blockIdx.x * 256 + threadIdx.x;
    if (idx >= total) return;
    const int    c   = (int)(idx % CONV_DIM);
    const size_t tok = idx / CONV_DIM;
    const int    l   = (int)(tok % SEQ);
    float acc = cb[c];
#pragma unroll
    for (int k = 0; k < D_CONV; ++k) {
        const int ls = l + k - (D_CONV - 1);
        if (ls >= 0)
            acc += zx[(tok + (size_t)(k - (D_CONV - 1))) * (size_t)E_TOTAL + D_INNER + c]
                   * cw[c * D_CONV + k];
    }
    out[idx] = silu_(acc);
}

// =====================================================================
// Kernel 5: dt = softplus(dt_raw + bias); dA = exp(-exp(A_log)*dt)
// =====================================================================
__global__ void __launch_bounds__(256)
dt_kernel(const float* __restrict__ zx, const float* __restrict__ dtb,
          const float* __restrict__ alog, float* __restrict__ dts,
          float* __restrict__ dAg, int total) {
    int idx = (int)blockIdx.x * 256 + threadIdx.x;
    if (idx >= total) return;
    const int    h   = idx & (NHEADS - 1);
    const size_t tok = (size_t)idx >> 6;
    const float  dtr = zx[tok * (size_t)E_TOTAL + (E_TOTAL - NHEADS) + h] + dtb[h];
    const float  sp  = softplus_(dtr);
    dts[idx] = sp;
    dAg[idx] = __expf(-__expf(alog[h]) * sp);
}

// =====================================================================
// Chunked selective scan. The recurrence S_l = dA_l*S_{l-1} + U_l has a
// scalar decay per (l,h), so with P_l = prod(dA) over the chunk prefix:
//     S_l = P_l * S_in  +  S_local_l
//     y_l = C_l . S_local_l  +  P_l * (C_l . S_in)
// Pass A: per-chunk local scan (parallel over b,h,chunk = 2048 blocks).
// Pass B: propagate incoming chunk states (128 blocks, 16 serial steps).
// Pass C: add the S_in correction + D*x (parallel, 2048 blocks).
// Thread mapping: t -> p = t>>2 (head dim), q = t&3 owns states n0..n0+15.
// =====================================================================
__global__ void __launch_bounds__(256)
scan_partial_kernel(const float* __restrict__ xbc, const float* __restrict__ dts,
                    const float* __restrict__ dAg, float* __restrict__ yloc,
                    float* __restrict__ Sbuf, float* __restrict__ Pend,
                    float* __restrict__ Ppre) {
    const int c = (int)blockIdx.x & (NCHUNK - 1);
    const int h = ((int)blockIdx.x >> 4) & (NHEADS - 1);
    const int b = (int)blockIdx.x >> 10;
    const int t = threadIdx.x;
    const int p  = t >> 2;
    const int q  = t & 3;
    const int n0 = q << 4;

    float s[16];
#pragma unroll
    for (int j = 0; j < 16; ++j) s[j] = 0.0f;
    float P = 1.0f;

    for (int i = 0; i < CHUNKL; ++i) {
        const int l = c * CHUNKL + i;
        const size_t tok = (size_t)b * SEQ + l;
        const float da = dAg[tok * NHEADS + h];
        const float dt = dts[tok * NHEADS + h];
        const float* base = xbc + tok * (size_t)CONV_DIM;
        const float xp = base[h * HEADDIM + p];
        const v4f* Bp = (const v4f*)(base + D_INNER + n0);
        const v4f* Cp = (const v4f*)(base + D_INNER + D_STATE + n0);
        const float coef = dt * xp;
        float acc = 0.0f;
#pragma unroll
        for (int j = 0; j < 4; ++j) {
            const v4f Bv = Bp[j];
            const v4f Cv = Cp[j];
            s[4 * j + 0] = s[4 * j + 0] * da + coef * Bv.x;  acc += s[4 * j + 0] * Cv.x;
            s[4 * j + 1] = s[4 * j + 1] * da + coef * Bv.y;  acc += s[4 * j + 1] * Cv.y;
            s[4 * j + 2] = s[4 * j + 2] * da + coef * Bv.z;  acc += s[4 * j + 2] * Cv.z;
            s[4 * j + 3] = s[4 * j + 3] * da + coef * Bv.w;  acc += s[4 * j + 3] * Cv.w;
        }
        P *= da;
        acc += __shfl_xor(acc, 1, 4);
        acc += __shfl_xor(acc, 2, 4);
        if (q == 0)
            yloc[tok * (size_t)D_INNER + h * HEADDIM + p] = acc;
        if (t == 0)
            Ppre[tok * NHEADS + h] = P;        // inclusive prefix product of dA
    }

    float* sb = Sbuf + (size_t)blockIdx.x * STATE_SZ + t * 16;
#pragma unroll
    for (int j = 0; j < 4; ++j) {
        v4f v; v.x = s[4*j+0]; v.y = s[4*j+1]; v.z = s[4*j+2]; v.w = s[4*j+3];
        ((v4f*)sb)[j] = v;
    }
    if (t == 0) Pend[blockIdx.x] = P;
}

__global__ void __launch_bounds__(256)
scan_prop_kernel(float* __restrict__ Sbuf, const float* __restrict__ Pend) {
    const int bh = blockIdx.x;            // b*NHEADS + h
    const int t  = threadIdx.x;
    float sin[16];
#pragma unroll
    for (int j = 0; j < 16; ++j) sin[j] = 0.0f;
    for (int c = 0; c < NCHUNK; ++c) {
        float* sb = Sbuf + ((size_t)bh * NCHUNK + c) * STATE_SZ + t * 16;
        v4f sl[4];
#pragma unroll
        for (int j = 0; j < 4; ++j) sl[j] = ((v4f*)sb)[j];
        // overwrite the local chunk-end state with the *incoming* state
#pragma unroll
        for (int j = 0; j < 4; ++j) {
            v4f v; v.x = sin[4*j+0]; v.y = sin[4*j+1]; v.z = sin[4*j+2]; v.w = sin[4*j+3];
            ((v4f*)sb)[j] = v;
        }
        const float pe = Pend[bh * NCHUNK + c];
#pragma unroll
        for (int j = 0; j < 4; ++j) {
            sin[4*j+0] = pe * sin[4*j+0] + sl[j].x;
            sin[4*j+1] = pe * sin[4*j+1] + sl[j].y;
            sin[4*j+2] = pe * sin[4*j+2] + sl[j].z;
            sin[4*j+3] = pe * sin[4*j+3] + sl[j].w;
        }
    }
}

__global__ void __launch_bounds__(256)
scan_fixup_kernel(const float* __restrict__ xbc, const float* __restrict__ Sbuf,
                  const float* __restrict__ Ppre, const float* __restrict__ Dp,
                  float* __restrict__ y) {
    const int c = (int)blockIdx.x & (NCHUNK - 1);
    const int h = ((int)blockIdx.x >> 4) & (NHEADS - 1);
    const int b = (int)blockIdx.x >> 10;
    const int t = threadIdx.x;
    const int p  = t >> 2;
    const int q  = t & 3;
    const int n0 = q << 4;

    const float* sb = Sbuf + (size_t)blockIdx.x * STATE_SZ + t * 16;
    float sin[16];
#pragma unroll
    for (int j = 0; j < 4; ++j) {
        const v4f v = ((const v4f*)sb)[j];
        sin[4*j+0] = v.x; sin[4*j+1] = v.y; sin[4*j+2] = v.z; sin[4*j+3] = v.w;
    }
    const float Dh = Dp[h];

    for (int i = 0; i < CHUNKL; ++i) {
        const int l = c * CHUNKL + i;
        const size_t tok = (size_t)b * SEQ + l;
        const float* base = xbc + tok * (size_t)CONV_DIM;
        const float xp = base[h * HEADDIM + p];
        const v4f* Cp = (const v4f*)(base + D_INNER + D_STATE + n0);
        float acc = 0.0f;
#pragma unroll
        for (int j = 0; j < 4; ++j) {
            const v4f Cv = Cp[j];
            acc += sin[4*j+0] * Cv.x + sin[4*j+1] * Cv.y
                 + sin[4*j+2] * Cv.z + sin[4*j+3] * Cv.w;
        }
        acc += __shfl_xor(acc, 1, 4);
        acc += __shfl_xor(acc, 2, 4);
        if (q == 0) {
            const float P = Ppre[tok * NHEADS + h];
            const size_t yi = tok * (size_t)D_INNER + h * HEADDIM + p;
            y[yi] = y[yi] + P * acc + Dh * xp;
        }
    }
}

// =====================================================================
// Kernel 7: gated RMSNorm: rmsnorm(y * silu(z)) * norm_w -> bf16
// =====================================================================
__global__ void __launch_bounds__(256)
gated_rmsnorm_kernel(const float* __restrict__ y, const float* __restrict__ zx,
                     const float* __restrict__ nw, unsigned short* __restrict__ out) {
    __shared__ float red[256];
    const size_t row = blockIdx.x;
    const float* yr = y  + row * (size_t)D_INNER;
    const float* zr = zx + row * (size_t)E_TOTAL;   // z = first D_INNER columns
    float ss = 0.0f;
    for (int i = threadIdx.x; i < D_INNER; i += 256) {
        const float g = yr[i] * silu_(zr[i]);
        ss += g * g;
    }
    red[threadIdx.x] = ss;
    __syncthreads();
    for (int s = 128; s > 0; s >>= 1) {
        if ((int)threadIdx.x < s) red[threadIdx.x] += red[threadIdx.x + s];
        __syncthreads();
    }
    const float scale = rsqrtf(red[0] / (float)D_INNER + EPS_RMS);
    unsigned short* orow = out + row * (size_t)D_INNER;
    for (int i = threadIdx.x; i < D_INNER; i += 256)
        orow[i] = f32_to_bf16_bits(yr[i] * silu_(zr[i]) * scale * nw[i]);
}

// =====================================================================
// Host launcher
// =====================================================================
extern "C" void kernel_launch(void* const* d_in, const int* in_sizes, int n_in,
                              void* d_out, int out_size, void* d_ws, size_t ws_size,
                              hipStream_t stream) {
    (void)in_sizes; (void)n_in; (void)out_size; (void)ws_size;

    const float* hidden = (const float*)d_in[0];  // [2,2048,2048]
    const float* rmsw   = (const float*)d_in[1];  // [2048]
    const float* inw    = (const float*)d_in[2];  // [8384,2048]
    const float* convw  = (const float*)d_in[3];  // [4224,4]
    const float* convb  = (const float*)d_in[4];  // [4224]
    const float* dtb    = (const float*)d_in[5];  // [64]
    const float* alog   = (const float*)d_in[6];  // [64]
    const float* Dp     = (const float*)d_in[7];  // [64]
    const float* normw  = (const float*)d_in[8];  // [4096]
    const float* outw   = (const float*)d_in[9];  // [2048,4096]
    float* out = (float*)d_out;                   // [2,2048,2048]

    // ---- workspace layout (256B aligned bump allocator) ----
    char* ws = (char*)d_ws;
    auto bump = [&](size_t bytes) -> char* {
        char* p = ws;
        ws += (bytes + 255) & ~(size_t)255;
        return p;
    };
    unsigned short* u16    = (unsigned short*)bump((size_t)NTOK * D_MODEL * 2);
    unsigned short* win16  = (unsigned short*)bump((size_t)E_TOTAL * D_MODEL * 2);
    unsigned short* wout16 = (unsigned short*)bump((size_t)D_MODEL * D_INNER * 2);
    float* zxbcdt          = (float*)bump((size_t)NTOK * E_TOTAL * 4);
    float* xbc             = (float*)bump((size_t)NTOK * CONV_DIM * 4);
    float* dts             = (float*)bump((size_t)NTOK * NHEADS * 4);
    float* dAg             = (float*)bump((size_t)NTOK * NHEADS * 4);
    float* yv              = (float*)bump((size_t)NTOK * D_INNER * 4);
    unsigned short* y16    = (unsigned short*)bump((size_t)NTOK * D_INNER * 2);
    float* Sbuf            = (float*)bump((size_t)BATCH * NHEADS * NCHUNK * STATE_SZ * 4);
    float* Pend            = (float*)bump((size_t)BATCH * NHEADS * NCHUNK * 4);
    float* Ppre            = (float*)bump((size_t)NTOK * NHEADS * 4);

    // 1. RMSNorm + bf16 convert of activations
    rmsnorm_bf16_kernel<<<NTOK, 256, 0, stream>>>(hidden, rmsw, u16, D_MODEL, EPS_RMS);

    // 2. Weight down-converts (fp32 -> bf16)
    {
        size_t n = (size_t)E_TOTAL * D_MODEL;
        f32_to_bf16_kernel<<<(unsigned)((n + 255) / 256), 256, 0, stream>>>(inw, win16, n);
    }
    {
        size_t n = (size_t)D_MODEL * D_INNER;
        f32_to_bf16_kernel<<<(unsigned)((n + 255) / 256), 256, 0, stream>>>(outw, wout16, n);
    }

    // 3. in_proj GEMM: zxbcdt[4096,8384] = u16[4096,2048] * win16[8384,2048]^T
    {
        const int M = NTOK, N = E_TOTAL, K = D_MODEL;
        const int waves = ((M + 63) / 64) * ((N + 31) / 32);
        gemm_bf16_wmma<<<(waves + 7) / 8, 256, 0, stream>>>(u16, win16, nullptr, zxbcdt, M, N, K);
    }

    // 4. depthwise causal conv + SiLU
    {
        const size_t total = (size_t)NTOK * CONV_DIM;
        conv_silu_kernel<<<(unsigned)((total + 255) / 256), 256, 0, stream>>>(
            zxbcdt, convw, convb, xbc, total);
    }

    // 5. dt softplus / dA
    {
        const int total = NTOK * NHEADS;
        dt_kernel<<<(total + 255) / 256, 256, 0, stream>>>(zxbcdt, dtb, alog, dts, dAg, total);
    }

    // 6. chunked selective scan (A: local, B: propagate, C: fixup)
    scan_partial_kernel<<<BATCH * NHEADS * NCHUNK, 256, 0, stream>>>(
        xbc, dts, dAg, yv, Sbuf, Pend, Ppre);
    scan_prop_kernel<<<BATCH * NHEADS, 256, 0, stream>>>(Sbuf, Pend);
    scan_fixup_kernel<<<BATCH * NHEADS * NCHUNK, 256, 0, stream>>>(
        xbc, Sbuf, Ppre, Dp, yv);

    // 7. gated RMSNorm -> bf16
    gated_rmsnorm_kernel<<<NTOK, 256, 0, stream>>>(yv, zxbcdt, normw, y16);

    // 8. out_proj GEMM + fused residual: out = y16 * wout16^T + hidden
    {
        const int M = NTOK, N = D_MODEL, K = D_INNER;
        const int waves = ((M + 63) / 64) * ((N + 31) / 32);
        gemm_bf16_wmma<<<(waves + 7) / 8, 256, 0, stream>>>(y16, wout16, hidden, out, M, N, K);
    }
}
